// RGCNDirectConv_70566312673745
// MI455X (gfx1250) — matrix operand order repaired and checked by reading, not compiled, not run on hardware
//
#include <hip/hip_runtime.h>
#include <stdint.h>

// ---------- types ----------
typedef __bf16 bf16;
typedef __attribute__((ext_vector_type(16))) __bf16 bf16x16;
typedef __attribute__((ext_vector_type(8)))  float  floatx8;

#define IN_F  128
#define OUT_F 128

// ---------------------------------------------------------------------------
// CDNA5 async global->LDS copy (no VGPR round trip, tracked by ASYNCcnt).
// VDST operand = LDS byte address (workgroup-relative), VADDR = 64-bit global.
// ---------------------------------------------------------------------------
__device__ __forceinline__ void async_copy_b128(unsigned ldsoff, uint64_t gaddr) {
    asm volatile("global_load_async_to_lds_b128 %0, %1, off"
                 :: "v"(ldsoff), "v"(gaddr) : "memory");
}
__device__ __forceinline__ void wait_async0() {
    asm volatile("s_wait_asynccnt 0x0" ::: "memory");
}

// ---------------------------------------------------------------------------
// 1) zero the bf16 aggregate (relations 0..R-1) -- required each call since
//    the scatter accumulates with atomics.
// ---------------------------------------------------------------------------
__global__ void zero_u4_kernel(uint4* __restrict__ p, size_t n) {
    size_t i      = (size_t)blockIdx.x * blockDim.x + threadIdx.x;
    size_t stride = (size_t)gridDim.x * blockDim.x;
    uint4 z = make_uint4(0u, 0u, 0u, 0u);
    for (; i < n; i += stride) p[i] = z;
}

// ---------------------------------------------------------------------------
// 2) x (f32) -> bf16 into aggregate slot R (self-loop treated as relation R)
// ---------------------------------------------------------------------------
__global__ void cvt_x_kernel(const float4* __restrict__ x4,
                             unsigned* __restrict__ dst, int nQuads) {
    int i = blockIdx.x * blockDim.x + threadIdx.x;
    if (i >= nQuads) return;
    float4 v = x4[i];
    union { bf16 h[2]; unsigned u; } p0, p1;
    p0.h[0] = (bf16)v.x; p0.h[1] = (bf16)v.y;
    p1.h[0] = (bf16)v.z; p1.h[1] = (bf16)v.w;
    dst[2 * i]     = p0.u;
    dst[2 * i + 1] = p1.u;
}

// ---------------------------------------------------------------------------
// 3) WrT[r][o][i] (bf16, transposed so K-pairs are contiguous for the WMMA
//    B-fragment): r<R: sum_b bw[r,b]*bases[b,i,o];  r==R: self_W[o,i].
// ---------------------------------------------------------------------------
__global__ void prep_wrt_kernel(const float* __restrict__ bw,
                                const float* __restrict__ bases,
                                const float* __restrict__ selfW,
                                unsigned short* __restrict__ wrt,
                                int R, int Bn, int total) {
    int idx = blockIdx.x * blockDim.x + threadIdx.x;
    if (idx >= total) return;
    int r   = idx >> 14;
    int rem = idx & 16383;
    int o   = rem >> 7;
    int i   = rem & 127;
    float s;
    if (r < R) {
        s = 0.0f;
        for (int b = 0; b < Bn; ++b)
            s += bw[r * Bn + b] * bases[((size_t)b * IN_F + i) * OUT_F + o];
    } else {
        s = selfW[o * IN_F + i];
    }
    union { bf16 h; unsigned short u; } c;
    c.h = (bf16)s;
    wrt[idx] = c.u;
}

// ---------------------------------------------------------------------------
// 4) edge scatter: one wave per edge; lane L handles features 4L..4L+3 and
//    issues two packed-bf16 global atomic adds (CDNA5 GLOBAL_ATOMIC_PK_ADD_BF16).
// ---------------------------------------------------------------------------
__global__ void scatter_edges_kernel(const float* __restrict__ x,
                                     const int*   __restrict__ eidx,
                                     const int*   __restrict__ etype,
                                     const float* __restrict__ norm,
                                     unsigned short* __restrict__ agg,
                                     int E, int nN) {
    long long gid = (long long)blockIdx.x * blockDim.x + threadIdx.x;
    int e    = (int)(gid >> 5);
    int lane = (int)(gid & 31);
    if (e >= E) return;

    int s = eidx[e];
    int d = eidx[E + e];
    int r = etype[e];
    float inv = 1.0f / norm[(size_t)r * nN + d];

    const float4* xr = (const float4*)(x + (size_t)s * IN_F);
    float4 v = xr[lane];
    v.x *= inv; v.y *= inv; v.z *= inv; v.w *= inv;

    union { bf16 h[2]; unsigned u; } p0, p1;
    p0.h[0] = (bf16)v.x; p0.h[1] = (bf16)v.y;
    p1.h[0] = (bf16)v.z; p1.h[1] = (bf16)v.w;

    unsigned short* dstp = agg + ((size_t)r * nN + d) * IN_F + lane * 4;
    uint64_t a0 = (uint64_t)dstp;
    uint64_t a1 = a0 + 4;
    asm volatile("global_atomic_pk_add_bf16 %0, %1, off"
                 :: "v"(a0), "v"(p0.u) : "memory");
    asm volatile("global_atomic_pk_add_bf16 %0, %1, off"
                 :: "v"(a1), "v"(p1.u) : "memory");
}

// ---------------------------------------------------------------------------
// 5) out[n,o] = sum_{r=0..R} agg_bf16[r,n,:] @ WrT[r,:,:]^T   (f32 accum)
//    Software-pipelined: 84 stages (r x 32-wide K chunk). Each stage async-
//    copies A(64x32) + B(128x32) bf16 chunks into the next LDS buffer while
//    the current stage's 4 WMMAs run. Rows padded to 80B (5 uint4) for banks.
//    LDS: 2 x (5120 + 10240) = 30720 B.
// ---------------------------------------------------------------------------
__global__ void __launch_bounds__(256)
rgcn_gemm_kernel(const unsigned short* __restrict__ agg,  // [(R+1), nN, 128] bf16
                 const unsigned short* __restrict__ wrt,  // [(R+1), 128, 128] bf16
                 float* __restrict__ out, int nN, int nRel) {
    __shared__ uint4 lds[1920];   // 2 buffers x (A: 64x5 | B: 128x5) uint4

    const int tid  = threadIdx.x;
    const int lane = tid & 31;
    const int wave = tid >> 5;
    const int nodeBase = blockIdx.x * 64;

    const int mBase  = (wave & 3) * 16;   // row strip within block
    const int nBaseW = (wave >> 2) * 64;  // 64-wide column group
    const int h  = lane >> 4;             // lane half (ISA 16-bit frag layout)
    const int ml = lane & 15;             // M (A) / N (B) index within tile

    // LDS byte base (generic address truncates to LDS offset per flat aperture)
    const unsigned ldsBase = (unsigned)(uintptr_t)&lds[0];

    // per-thread stage-copy geometry (aQ*16B within a 64B chunk row)
    const int aRow  = tid >> 2;            // 0..63
    const int aQ    = tid & 3;             // 0..3
    int aNode = nodeBase + aRow;
    if (aNode >= nN) aNode = nN - 1;       // clamp; masked at store time
    const int bRow0 = tid >> 2;            // 0..63
    const int bRow1 = bRow0 + 64;          // 64..127

    floatx8 acc[4];
#pragma unroll
    for (int t = 0; t < 4; ++t) acc[t] = (floatx8)0.0f;

    const int S = nRel * 4;                // stages: (r, kb)

    auto issue_stage = [&](int s, int buf) {
        const int r  = s >> 2;
        const int kb = (s & 3) << 5;                    // 0,32,64,96
        const unsigned aBase = ldsBase + (unsigned)buf * 15360u;
        const unsigned bBase = aBase + 5120u;
        // A chunk: agg[r, node, kb..kb+31]
        uint64_t ga = (uint64_t)(uintptr_t)(agg + ((size_t)r * nN + aNode) * IN_F + kb)
                      + (unsigned)(aQ * 16);
        async_copy_b128(aBase + (unsigned)(aRow * 80 + aQ * 16), ga);
        // B chunk: wrt[r, o, kb..kb+31] for o = 0..127
        const unsigned short* wb = wrt + (size_t)r * OUT_F * IN_F + kb;
        uint64_t gb0 = (uint64_t)(uintptr_t)(wb + (size_t)bRow0 * IN_F) + (unsigned)(aQ * 16);
        uint64_t gb1 = (uint64_t)(uintptr_t)(wb + (size_t)bRow1 * IN_F) + (unsigned)(aQ * 16);
        async_copy_b128(bBase + (unsigned)(bRow0 * 80 + aQ * 16), gb0);
        async_copy_b128(bBase + (unsigned)(bRow1 * 80 + aQ * 16), gb1);
    };

    auto compute_stage = [&](int buf) {
        const int aB = buf * 960;          // uint4 index of A chunk
        const int bB = aB + 320;           // uint4 index of B chunk
        union { uint4 u[2]; bf16x16 v; } afrag;
        const int arow = mBase + ml;
        // A 16x32 frag: lanes 0-15 K=0..7,16..23 ; lanes 16-31 K=8..15,24..31
        afrag.u[0] = lds[aB + arow * 5 + h];
        afrag.u[1] = lds[aB + arow * 5 + 2 + h];
#pragma unroll
        for (int t = 0; t < 4; ++t) {
            union { uint4 u[2]; bf16x16 v; } bfrag;
            const int orow = nBaseW + t * 16 + ml;
            // B 32x16 frag: lanes 0-15 K=0..7,16..23 ; lanes 16-31 K=8..15,24..31
            bfrag.u[0] = lds[bB + orow * 5 + h];
            bfrag.u[1] = lds[bB + orow * 5 + 2 + h];
            acc[t] = __builtin_amdgcn_wmma_f32_16x16x32_bf16(
                false, afrag.v, false, bfrag.v,
                (short)0, acc[t], false, false);
        }
    };

    // prologue: stage 0 into buffer 0
    issue_stage(0, 0);
    wait_async0();
    __syncthreads();

    for (int s = 0; s < S; ++s) {
        if (s + 1 < S) issue_stage(s + 1, (s + 1) & 1);   // prefetch next
        compute_stage(s & 1);                              // overlap WMMAs
        wait_async0();                                     // own prefetch done
        __syncthreads();                                   // everyone's done
    }

    // store: C/D layout: lanes 0-15 -> M=vgpr, lanes 16-31 -> M=vgpr+8
#pragma unroll
    for (int t = 0; t < 4; ++t) {
        int col = nBaseW + t * 16 + ml;
#pragma unroll
        for (int v = 0; v < 8; ++v) {
            int m   = (h == 0) ? v : (v + 8);
            int row = nodeBase + mBase + m;
            if (row < nN) out[(size_t)row * OUT_F + col] = acc[t][v];
        }
    }
}

// ---------------------------------------------------------------------------
// host wrapper
// ---------------------------------------------------------------------------
extern "C" void kernel_launch(void* const* d_in, const int* in_sizes, int n_in,
                              void* d_out, int out_size, void* d_ws, size_t ws_size,
                              hipStream_t stream) {
    const float* x     = (const float*)d_in[0];
    const int*   eidx  = (const int*)d_in[1];
    const int*   etype = (const int*)d_in[2];
    const float* norm  = (const float*)d_in[3];
    const float* selfW = (const float*)d_in[4];
    const float* bases = (const float*)d_in[5];
    const float* bw    = (const float*)d_in[6];
    float* out = (float*)d_out;

    const int E  = in_sizes[2];
    const int nN = in_sizes[0] / IN_F;
    const int R  = in_sizes[3] / nN;
    const int Bn = in_sizes[6] / R;

    // workspace layout: [ agg bf16 (R+1, nN, 128) | WrT bf16 (R+1, 128, 128) ]
    unsigned short* agg = (unsigned short*)d_ws;
    size_t aggBytes = (size_t)(R + 1) * nN * IN_F * 2;
    size_t wrtOff   = (aggBytes + 255) & ~(size_t)255;
    unsigned short* wrt = (unsigned short*)((char*)d_ws + wrtOff);

    // 1) zero relation slots 0..R-1
    size_t zeroQ = (size_t)R * nN * (IN_F * 2 / 16);
    zero_u4_kernel<<<4096, 256, 0, stream>>>((uint4*)agg, zeroQ);

    // 2) x -> bf16 into slot R
    int nQuads = nN * (IN_F / 4);
    unsigned* xdst = (unsigned*)(agg + (size_t)R * nN * IN_F);
    cvt_x_kernel<<<(nQuads + 255) / 256, 256, 0, stream>>>(
        (const float4*)x, xdst, nQuads);

    // 3) relation weights (transposed, bf16)
    int totalW = (R + 1) * OUT_F * IN_F;
    prep_wrt_kernel<<<(totalW + 255) / 256, 256, 0, stream>>>(
        bw, bases, selfW, wrt, R, Bn, totalW);

    // 4) edge scatter (one wave per edge)
    long long sThreads = (long long)E * 32;
    int sBlocks = (int)((sThreads + 255) / 256);
    scatter_edges_kernel<<<sBlocks, 256, 0, stream>>>(
        x, eidx, etype, norm, agg, E, nN);

    // 5) batched bf16 WMMA GEMM over (R+1) relations, async-pipelined
    int gBlocks = (nN + 63) / 64;
    rgcn_gemm_kernel<<<gBlocks, 256, 0, stream>>>(agg, wrt, out, nN, R + 1);
}
